// SparseCoder_90838558310778
// MI455X (gfx1250) — matrix operand order repaired and checked by reading, not compiled, not run on hardware
//
#include <hip/hip_runtime.h>
#include <hip/hip_bf16.h>
#include <math.h>

// ---------------- problem constants ----------------
#define EMBED_DIM   1024
#define BASIS_DIM   2048
#define BATCH       128
#define LAMB        0.3f
#define N_ITERS     500
#define H_STEP      0.005f
#define RM_CONST    0.99f        // (HISTORY_LEN-1)/HISTORY_LEN
#define INV_HIST    0.01f        // 1/HISTORY_LEN
#define LOWEST_ACT  0.001f
#define POWER_ITERS 25

// ---------------- GEMM tiling ----------------
#define BLK  64    // output tile per workgroup (64x64)
#define TK   16    // K-tile for fp32 16x16x4 path
#define TKB  64    // K-tile for split-bf16 path (two 16x16x32 k-steps)
#define TKP  (TKB + 8)  // padded LDS row stride: 72 halves = 144B (16B multiple)

typedef float  v2f  __attribute__((ext_vector_type(2)));
typedef float  v8f  __attribute__((ext_vector_type(8)));
typedef __bf16 bf16_t;
typedef __bf16 v8bf  __attribute__((ext_vector_type(8)));
typedef __bf16 v16bf __attribute__((ext_vector_type(16)));

enum { MODE_STORE = 0, MODE_ISTA = 1, MODE_RES = 2, MODE_DB = 3 };

// ======================================================================
// Exact-fp32 WMMA GEMM (V_WMMA_F32_16X16X4_F32) — used for the four
// one-shot GEMMs where exactness matters (G, c, res, dB).
// ======================================================================
__device__ __forceinline__ void epilogue(int mode, int m, int n, int M, int N, float t,
                                         float* __restrict__ out,
                                         const float* __restrict__ aux0,
                                         const float* __restrict__ aux1,
                                         float* __restrict__ colsq)
{
    long o = (long)m * N + n;
    if (mode == MODE_STORE) {
        out[o] = t;
    } else if (mode == MODE_RES) {
        out[o] = aux0[(long)n * M + m] - t;      // I_t[d][b] = I[b*D + d]
    } else { // MODE_DB
        float val = aux0[o] + (t * (H_STEP / (float)BATCH)) / aux1[n];
        out[o] = val;
        atomicAdd(&colsq[n], val * val);
    }
}

__global__ __launch_bounds__(256) void wmma_gemm_f32(
    const float* __restrict__ A, long sAm, long sAk,
    const float* __restrict__ B, long sBk, long sBn,
    float* __restrict__ out, int M, int N, int K, int mode,
    const float* __restrict__ aux0, const float* __restrict__ aux1,
    float* __restrict__ colsq)
{
    __shared__ float As[BLK][TK];
    __shared__ float Bs[TK][BLK];

    const int tid  = threadIdx.x;
    const int lane = tid & 31;
    const int wave = tid >> 5;
    const int wrow = wave & 3;
    const int wcol = wave >> 2;
    const int bm = blockIdx.x * BLK;
    const int bn = blockIdx.y * BLK;

    v8f acc0 = {};
    v8f acc1 = {};

    const int lrow  = lane & 15;
    const int lksel = (lane >> 4) * 2;

    for (int k0 = 0; k0 < K; k0 += TK) {
        for (int i = tid; i < BLK * TK; i += 256) {
            int m  = i >> 4;
            int kk = i & 15;
            As[m][kk] = A[(long)(bm + m) * sAm + (long)(k0 + kk) * sAk];
        }
        for (int i = tid; i < TK * BLK; i += 256) {
            int kk = i >> 6;
            int n  = i & 63;
            Bs[kk][n] = B[(long)(k0 + kk) * sBk + (long)(bn + n) * sBn];
        }
        // warm near caches with next K-stage while this one computes
        if (k0 + TK < K) {
            int m  = tid >> 4, kk = tid & 15;
            __builtin_prefetch(&A[(long)(bm + m) * sAm + (long)(k0 + TK + kk) * sAk], 0, 3);
            int k2 = tid >> 6, n = tid & 63;
            __builtin_prefetch(&B[(long)(k0 + TK + k2) * sBk + (long)(bn + n) * sBn], 0, 3);
        }
        __syncthreads();

        const int arow  = wrow * 16 + lrow;
        const int bcol0 = wcol * 32 + lrow;
#pragma unroll
        for (int kk = 0; kk < TK; kk += 4) {
            v2f af, bf0, bf1;
            af.x  = As[arow][kk + lksel];
            af.y  = As[arow][kk + lksel + 1];
            bf0.x = Bs[kk + lksel][bcol0];
            bf0.y = Bs[kk + lksel + 1][bcol0];
            bf1.x = Bs[kk + lksel][bcol0 + 16];
            bf1.y = Bs[kk + lksel + 1][bcol0 + 16];
            acc0 = __builtin_amdgcn_wmma_f32_16x16x4_f32(false, af, false, bf0,
                                                         (short)0, acc0, false, false);
            acc1 = __builtin_amdgcn_wmma_f32_16x16x4_f32(false, af, false, bf1,
                                                         (short)0, acc1, false, false);
        }
        __syncthreads();
    }

    const int mG = bm + wrow * 16 + ((lane >> 4) << 3);
    const int nG = bn + wcol * 32 + (lane & 15);
#pragma unroll
    for (int r = 0; r < 8; ++r) {
        epilogue(mode, mG + r, nG,      M, N, acc0[r], out, aux0, aux1, colsq);
        epilogue(mode, mG + r, nG + 16, M, N, acc1[r], out, aux0, aux1, colsq);
    }
}

// ======================================================================
// Split-bf16 ISTA GEMM (V_WMMA_F32_16X16X32_BF16, bf16x3 ~ fp32):
//   t = G a ;  a_new = softshrink(a_old + eta*(c - t), thr)
// Sources are pre-split bf16 hi/lo (G row-major, a^T row-major), epilogue
// emits fp32 a_new plus its bf16 hi/lo transposed copies for the next step.
// Staging moves aligned 16B v8bf chunks (b128 loads / b128 LDS stores).
// ======================================================================
__device__ __forceinline__ v16bf load_fragA(const bf16_t* rowp, int s) {
    // A 16x32 bf16 fragment: halves K[s..s+7] ++ K[s+16..s+23], s = 8*(lane>=16)
    const v8bf* p0 = (const v8bf*)(rowp + s);
    const v8bf* p1 = (const v8bf*)(rowp + s + 16);
    return __builtin_shufflevector(*p0, *p1, 0,1,2,3,4,5,6,7,8,9,10,11,12,13,14,15);
}
__device__ __forceinline__ v16bf load_fragB(const bf16_t* colp, int s) {
    // B 32x16 bf16 fragment: halves K[s..s+7] ++ K[s+16..s+23], s = 8*(lane>=16); LDS tile is col-major
    const v8bf* p0 = (const v8bf*)(colp + s);
    const v8bf* p1 = (const v8bf*)(colp + s + 16);
    return __builtin_shufflevector(*p0, *p1, 0,1,2,3,4,5,6,7,8,9,10,11,12,13,14,15);
}
#define WMMA_BF16(A_, B_, C_) \
    __builtin_amdgcn_wmma_f32_16x16x32_bf16(false, (A_), false, (B_), (short)0, (C_), false, false)

__global__ __launch_bounds__(256) void ista_wmma_gemm_bf16x3(
    const bf16_t* __restrict__ Ghi, const bf16_t* __restrict__ Glo,   // [2048,2048]
    const bf16_t* __restrict__ aThi, const bf16_t* __restrict__ aTlo, // [128,2048] (a^T)
    const float* __restrict__ a_old,  // [2048,128]
    const float* __restrict__ c,      // [2048,128]
    const float* __restrict__ scal,   // [eta, thr]
    float* __restrict__ a_new,        // [2048,128]
    bf16_t* __restrict__ oThi, bf16_t* __restrict__ oTlo) // [128,2048]
{
    __shared__ bf16_t Ah[BLK][TKP];
    __shared__ bf16_t Al[BLK][TKP];
    __shared__ bf16_t Bh[BLK][TKP];   // col-major: [n][k]
    __shared__ bf16_t Bl[BLK][TKP];

    const int tid  = threadIdx.x;
    const int lane = tid & 31;
    const int wave = tid >> 5;
    const int wrow = wave & 3;
    const int wcol = wave >> 2;
    const int bm = blockIdx.x * BLK;
    const int bn = blockIdx.y * BLK;

    v8f acc0 = {};
    v8f acc1 = {};

    const int arow  = wrow * 16 + (lane & 15);
    const int bcol0 = wcol * 32 + (lane & 15);
    const int sa = (lane >> 4) * 8;
    const int sb = (lane >> 4) * 8;

    // per-thread 16B chunk coordinates for staging: 64 rows x 8 chunks = 512
    for (int k0 = 0; k0 < BASIS_DIM; k0 += TKB) {
#pragma unroll
        for (int j = 0; j < 2; ++j) {
            int idx = tid + j * 256;
            int r   = idx >> 3;
            int ck  = (idx & 7) * 8;
            long ga = (long)(bm + r) * BASIS_DIM + (k0 + ck);
            long gb = (long)(bn + r) * BASIS_DIM + (k0 + ck);
            *(v8bf*)&Ah[r][ck] = *(const v8bf*)&Ghi[ga];
            *(v8bf*)&Al[r][ck] = *(const v8bf*)&Glo[ga];
            *(v8bf*)&Bh[r][ck] = *(const v8bf*)&aThi[gb];
            *(v8bf*)&Bl[r][ck] = *(const v8bf*)&aTlo[gb];
        }
        // prefetch next K-stage (global_prefetch_b8; no counters, pure warming)
        if (k0 + TKB < BASIS_DIM) {
            int r  = tid >> 3;
            int ck = (tid & 7) * 8;
            long ga = (long)(bm + r) * BASIS_DIM + (k0 + TKB + ck);
            long gb = (long)(bn + r) * BASIS_DIM + (k0 + TKB + ck);
            __builtin_prefetch(&Ghi[ga], 0, 3);
            __builtin_prefetch(&Glo[ga], 0, 3);
            __builtin_prefetch(&aThi[gb], 0, 3);
            __builtin_prefetch(&aTlo[gb], 0, 3);
        }
        __syncthreads();

#pragma unroll
        for (int ks = 0; ks < TKB; ks += 32) {
            v16bf ah  = load_fragA(&Ah[arow][ks], sa);
            v16bf al  = load_fragA(&Al[arow][ks], sa);
            v16bf bh0 = load_fragB(&Bh[bcol0][ks], sb);
            v16bf bl0 = load_fragB(&Bl[bcol0][ks], sb);
            v16bf bh1 = load_fragB(&Bh[bcol0 + 16][ks], sb);
            v16bf bl1 = load_fragB(&Bl[bcol0 + 16][ks], sb);

            acc0 = WMMA_BF16(ah, bh0, acc0);   // hi*hi
            acc1 = WMMA_BF16(ah, bh1, acc1);
            acc0 = WMMA_BF16(ah, bl0, acc0);   // hi*lo
            acc1 = WMMA_BF16(ah, bl1, acc1);
            acc0 = WMMA_BF16(al, bh0, acc0);   // lo*hi
            acc1 = WMMA_BF16(al, bh1, acc1);
        }
        __syncthreads();
    }

    const float eta = scal[0], thr = scal[1];
    const int mG = bm + wrow * 16 + ((lane >> 4) << 3);
    const int nG = bn + wcol * 32 + (lane & 15);
#pragma unroll
    for (int r = 0; r < 8; ++r) {
#pragma unroll
        for (int half = 0; half < 2; ++half) {
            int m = mG + r;
            int n = nG + 16 * half;
            float t = half ? acc1[r] : acc0[r];
            long o = (long)m * BATCH + n;
            float x   = a_old[o] + eta * (c[o] - t);
            float mag = fabsf(x) - thr;
            float v   = (mag > 0.0f) ? copysignf(mag, x) : 0.0f;
            a_new[o] = v;
            long oT = (long)n * BASIS_DIM + m;
            bf16_t h = (bf16_t)v;
            oThi[oT] = h;
            oTlo[oT] = (bf16_t)(v - (float)h);
        }
    }
}

// ---------------- small helper kernels ----------------
__global__ void zero_kernel(float* p, int n) {
    int i = blockIdx.x * blockDim.x + threadIdx.x;
    if (i < n) p[i] = 0.0f;
}
__global__ void ones_kernel(float* p, int n) {
    int i = blockIdx.x * blockDim.x + threadIdx.x;
    if (i < n) p[i] = 1.0f;
}
// split fp32 -> bf16 hi/lo
__global__ void cvt_split_kernel(const float* __restrict__ x,
                                 bf16_t* __restrict__ hi, bf16_t* __restrict__ lo, int n) {
    int i = blockIdx.x * blockDim.x + threadIdx.x;
    if (i < n) {
        float v = x[i];
        bf16_t h = (bf16_t)v;
        hi[i] = h;
        lo[i] = (bf16_t)(v - (float)h);
    }
}
// u = G * v  (G is n x n row-major)
__global__ void mv_kernel(const float* __restrict__ G, const float* __restrict__ v,
                          float* __restrict__ u, int n) {
    int r = blockIdx.x * blockDim.x + threadIdx.x;
    if (r >= n) return;
    const float* row = G + (long)r * n;
    float s = 0.0f;
    for (int k = 0; k < n; ++k) s = fmaf(row[k], v[k], s);
    u[r] = s;
}
__global__ void sumsq_kernel(const float* __restrict__ x, int n, float* acc) {
    __shared__ float s[256];
    float t = 0.0f;
    for (int i = blockIdx.x * 256 + threadIdx.x; i < n; i += gridDim.x * 256) {
        float v = x[i]; t = fmaf(v, v, t);
    }
    s[threadIdx.x] = t; __syncthreads();
    for (int o = 128; o > 0; o >>= 1) {
        if (threadIdx.x < o) s[threadIdx.x] += s[threadIdx.x + o];
        __syncthreads();
    }
    if (threadIdx.x == 0) atomicAdd(acc, s[0]);
}
__global__ void scale_kernel(const float* __restrict__ u, float* __restrict__ v,
                             int n, const float* __restrict__ acc) {
    int i = blockIdx.x * blockDim.x + threadIdx.x;
    if (i < n) v[i] = u[i] * rsqrtf(fmaxf(acc[0], 1e-30f));
}
__global__ void finalize_eta_kernel(const float* __restrict__ acc, float* scal) {
    float lam = sqrtf(fmaxf(acc[0], 1e-30f));   // lambda_max(G) ~ ||G v||, v unit
    scal[0] = 1.0f / lam;                       // eta
    scal[1] = LAMB / lam;                       // threshold
}
__global__ void stats_kernel(const float* __restrict__ a,
                             const float* __restrict__ hes_in,
                             const float* __restrict__ l1_in,
                             float* __restrict__ hes_out,
                             float* __restrict__ l1_out,
                             float* __restrict__ hes_eps) {
    int m = blockIdx.x * blockDim.x + threadIdx.x;
    if (m >= BASIS_DIM) return;
    const float* row = a + (long)m * BATCH;
    float s1 = 0.0f, s2 = 0.0f;
    for (int b = 0; b < BATCH; ++b) {
        float v = row[b];
        s1 += fabsf(v);
        s2 = fmaf(v, v, s2);
    }
    float nh = hes_in[m] * RM_CONST + (s2 / (float)BATCH) * INV_HIST;
    l1_out[m]  = l1_in[m] * RM_CONST + (s1 / (float)BATCH) * INV_HIST;
    hes_out[m] = nh;
    hes_eps[m] = nh + LOWEST_ACT;
}
__global__ void normalize_kernel(float* __restrict__ nb, const float* __restrict__ colsq, int n) {
    int i = blockIdx.x * blockDim.x + threadIdx.x;
    if (i < n) {
        int m = i & (BASIS_DIM - 1);
        nb[i] = nb[i] / sqrtf(fmaxf(colsq[m], 1e-30f));
    }
}
__global__ void snr_kernel(const float* __restrict__ acc,
                           const float* __restrict__ sig_in,
                           const float* __restrict__ noise_in,
                           float* __restrict__ out_snr) {
    float new_sig   = sig_in[0] * RM_CONST + acc[0] * INV_HIST;
    float new_noise = noise_in[0] * RM_CONST;
    out_snr[0] = new_sig / new_noise;
}

// ---------------- driver ----------------
extern "C" void kernel_launch(void* const* d_in, const int* in_sizes, int n_in,
                              void* d_out, int out_size, void* d_ws, size_t ws_size,
                              hipStream_t stream) {
    const float* I     = (const float*)d_in[0];   // [128,1024]
    const float* basis = (const float*)d_in[1];   // [1024,2048]
    const float* hes_i = (const float*)d_in[2];   // [2048]
    const float* l1_i  = (const float*)d_in[3];   // [2048]
    const float* sig_i = (const float*)d_in[4];   // scalar
    const float* noi_i = (const float*)d_in[5];   // scalar
    (void)in_sizes; (void)n_in; (void)out_size; (void)ws_size;

    // d_out layout: a_hat | new_basis | new_hes | new_l1 | snr
    float* out_a     = (float*)d_out;
    float* out_basis = out_a + (long)BASIS_DIM * BATCH;
    float* out_hes   = out_basis + (long)EMBED_DIM * BASIS_DIM;
    float* out_l1    = out_hes + BASIS_DIM;
    float* out_snr   = out_l1 + BASIS_DIM;

    // workspace carve-up (fp32 region, then bf16 region)
    float* ws      = (float*)d_ws;
    float* G       = ws;                                 // 2048*2048
    float* Cm      = G    + (long)BASIS_DIM * BASIS_DIM; // 2048*128
    float* bufA    = Cm   + (long)BASIS_DIM * BATCH;
    float* bufB    = bufA + (long)BASIS_DIM * BATCH;
    float* resbuf  = bufB + (long)BASIS_DIM * BATCH;     // 1024*128
    float* hes_eps = resbuf + (long)EMBED_DIM * BATCH;   // 2048
    float* colsq   = hes_eps + BASIS_DIM;                // 2048
    float* pv      = colsq + BASIS_DIM;                  // 2048
    float* pu      = pv + BASIS_DIM;                     // 2048
    float* scal    = pu + BASIS_DIM;                     // [eta, thr] (+pad)
    float* acc1    = scal + 4;
    float* acc2    = acc1 + 1;
    float* fend    = acc2 + 3;                           // 16B-align the bf16 region
    bf16_t* Ghi   = (bf16_t*)fend;                        // 2048*2048 halves
    bf16_t* Glo   = Ghi + (long)BASIS_DIM * BASIS_DIM;
    bf16_t* aThiA = Glo + (long)BASIS_DIM * BASIS_DIM;    // [128,2048] halves each
    bf16_t* aTloA = aThiA + (long)BATCH * BASIS_DIM;
    bf16_t* aThiB = aTloA + (long)BATCH * BASIS_DIM;
    bf16_t* aTloB = aThiB + (long)BATCH * BASIS_DIM;

    const int NA = BASIS_DIM * BATCH;

    // a0 = 0 (fp32 and both bf16 halves of the ping buffer)
    zero_kernel<<<(NA + 255) / 256, 256, 0, stream>>>(bufA, NA);
    zero_kernel<<<(NA / 2 + 255) / 256, 256, 0, stream>>>((float*)aThiA, NA / 2);
    zero_kernel<<<(NA / 2 + 255) / 256, 256, 0, stream>>>((float*)aTloA, NA / 2);

    // G = B^T B (exact fp32 matrix path), then split to bf16 hi/lo once
    wmma_gemm_f32<<<dim3(BASIS_DIM / BLK, BASIS_DIM / BLK), 256, 0, stream>>>(
        basis, 1L, (long)BASIS_DIM,  basis, (long)BASIS_DIM, 1L,
        G, BASIS_DIM, BASIS_DIM, EMBED_DIM, MODE_STORE, nullptr, nullptr, nullptr);
    cvt_split_kernel<<<(BASIS_DIM * BASIS_DIM) / 256, 256, 0, stream>>>(
        G, Ghi, Glo, BASIS_DIM * BASIS_DIM);

    // c = B^T I_t   (I_t[k][n] = I[n*1024 + k])
    wmma_gemm_f32<<<dim3(BASIS_DIM / BLK, BATCH / BLK), 256, 0, stream>>>(
        basis, 1L, (long)BASIS_DIM,  I, 1L, (long)EMBED_DIM,
        Cm, BASIS_DIM, BATCH, EMBED_DIM, MODE_STORE, nullptr, nullptr, nullptr);

    // power iteration on G for L = lambda_max
    ones_kernel<<<(BASIS_DIM + 255) / 256, 256, 0, stream>>>(pv, BASIS_DIM);
    for (int p = 0; p < POWER_ITERS; ++p) {
        mv_kernel<<<BASIS_DIM / 256, 256, 0, stream>>>(G, pv, pu, BASIS_DIM);
        zero_kernel<<<1, 1, 0, stream>>>(acc1, 1);
        sumsq_kernel<<<8, 256, 0, stream>>>(pu, BASIS_DIM, acc1);
        scale_kernel<<<BASIS_DIM / 256, 256, 0, stream>>>(pu, pv, BASIS_DIM, acc1);
    }
    finalize_eta_kernel<<<1, 1, 0, stream>>>(acc1, scal);

    // 500 ISTA steps on the bf16x3 matrix pipe
    for (int it = 1; it <= N_ITERS; ++it) {
        const bool odd = (it & 1) != 0;
        const float*  a_old = odd ? bufA : bufB;
        const bf16_t* ihi   = odd ? aThiA : aThiB;
        const bf16_t* ilo   = odd ? aTloA : aTloB;
        float*  a_new = (it == N_ITERS) ? out_a : (odd ? bufB : bufA);
        bf16_t* ohi   = odd ? aThiB : aThiA;
        bf16_t* olo   = odd ? aTloB : aTloA;
        ista_wmma_gemm_bf16x3<<<dim3(BASIS_DIM / BLK, BATCH / BLK), 256, 0, stream>>>(
            Ghi, Glo, ihi, ilo, a_old, Cm, scal, a_new, ohi, olo);
    }

    // running-mean stats (also builds hessian preconditioner)
    stats_kernel<<<BASIS_DIM / 256, 256, 0, stream>>>(out_a, hes_i, l1_i,
                                                      out_hes, out_l1, hes_eps);

    // res = I_t - B a   (exact fp32 path)
    wmma_gemm_f32<<<dim3(EMBED_DIM / BLK, BATCH / BLK), 256, 0, stream>>>(
        basis, (long)BASIS_DIM, 1L,  out_a, (long)BATCH, 1L,
        resbuf, EMBED_DIM, BATCH, BASIS_DIM, MODE_RES, I, nullptr, nullptr);

    // new_basis(unnorm) = basis + (H/B)*(res a^T)/hes_eps ; accumulate column sumsq
    zero_kernel<<<(BASIS_DIM + 255) / 256, 256, 0, stream>>>(colsq, BASIS_DIM);
    wmma_gemm_f32<<<dim3(EMBED_DIM / BLK, BASIS_DIM / BLK), 256, 0, stream>>>(
        resbuf, (long)BATCH, 1L,  out_a, 1L, (long)BATCH,
        out_basis, EMBED_DIM, BASIS_DIM, BATCH, MODE_DB, basis, hes_eps, colsq);

    // L2-normalize columns
    normalize_kernel<<<(EMBED_DIM * BASIS_DIM) / 256, 256, 0, stream>>>(
        out_basis, colsq, EMBED_DIM * BASIS_DIM);

    // snr
    zero_kernel<<<1, 1, 0, stream>>>(acc2, 1);
    sumsq_kernel<<<64, 256, 0, stream>>>(I, BATCH * EMBED_DIM, acc2);
    snr_kernel<<<1, 1, 0, stream>>>(acc2, sig_i, noi_i, out_snr);
}
